// FeatureEmbedding_77283641524369
// MI455X (gfx1250) — matrix operand order repaired and checked by reading, not compiled, not run on hardware
//
#include <hip/hip_runtime.h>
#include <hip/hip_bf16.h>
#include <stdint.h>

// ---------------------------------------------------------------------------
// FeatureEmbedding: out[b, 1+f, e] = relu(x[b,f]*W1[f,:]+b1[f,:]) @ W2[f] + b2[f]
//                   out[b, 0,   e] = cls_token[e]
// B=4096, F=64, D=256. Store-bandwidth bound (~273MB out @ 23.3TB/s ~ 12us).
// bf16 WMMA compute, async-to-LDS double-buffered B staging.
// ---------------------------------------------------------------------------

#define NF 64
#define DM 256
#define KC 32   // K chunk per WMMA step

typedef __attribute__((ext_vector_type(16))) __bf16 v16bf;
typedef __attribute__((ext_vector_type(8)))  float  v8f;

__device__ __forceinline__ uint32_t f32_to_bf16(float f) {
    uint32_t u = __builtin_bit_cast(uint32_t, f);
    u += 0x7FFFu + ((u >> 16) & 1u);   // round-to-nearest-even
    return u >> 16;
}

// ---------------------------------------------------------------------------
// Prep: W2[f][d][e] (f32) -> W2T[f][e][d] (bf16, N-major) via LDS tile
// transpose so both global read and write are coalesced.
// grid: (DM/32, DM/32, NF), block: (32, 8)
// ---------------------------------------------------------------------------
__global__ __launch_bounds__(256) void w2_transpose_bf16(
    const float* __restrict__ W2, uint16_t* __restrict__ W2T)
{
    __shared__ float tile[32][33];
    const int f  = blockIdx.z;
    const int tx = threadIdx.x, ty = threadIdx.y;
    const size_t base = (size_t)f * DM * DM;

    #pragma unroll
    for (int i = 0; i < 4; ++i) {
        int d = blockIdx.y * 32 + ty + 8 * i;
        int e = blockIdx.x * 32 + tx;
        tile[ty + 8 * i][tx] = W2[base + (size_t)d * DM + e];
    }
    __syncthreads();
    #pragma unroll
    for (int i = 0; i < 4; ++i) {
        int e = blockIdx.x * 32 + ty + 8 * i;
        int d = blockIdx.y * 32 + tx;
        W2T[base + (size_t)e * DM + d] = (uint16_t)f32_to_bf16(tile[tx][ty + 8 * i]);
    }
}

// ---------------------------------------------------------------------------
// cls rows: out[b, 0, :] = cls_token
// ---------------------------------------------------------------------------
__global__ __launch_bounds__(256) void cls_fill(
    const float* __restrict__ cls, float* __restrict__ out, int nB)
{
    int idx = blockIdx.x * 256 + threadIdx.x;
    if (idx >= nB * DM) return;
    int b = idx >> 8;
    int e = idx & (DM - 1);
    out[(size_t)b * (NF + 1) * DM + e] = cls[e];
}

// ---------------------------------------------------------------------------
// Main GEMM. grid: (B/128, NF), block: 512 threads (16 waves).
// Wave w: rows m0 = bx*128 + (w>>1)*16 .. +15, cols (w&1)*128 .. +127
// -> 8 accumulator tiles of 16x16 f32 (64 acc VGPRs; high occupancy).
// B (W2T, N-major bf16) staged per 32-K chunk into LDS with
// global_load_async_to_lds_b128 (ASYNCcnt), double-buffered.
// A-fragment built in-register from relu(x*W1+b1) per the 16-bit A layout.
// ---------------------------------------------------------------------------
__global__ __launch_bounds__(512) void feat_gemm(
    const float* __restrict__ x, const float* __restrict__ W1,
    const float* __restrict__ b1, const uint16_t* __restrict__ W2T,
    const float* __restrict__ b2, float* __restrict__ out)
{
    const int f     = blockIdx.y;
    const int tid   = threadIdx.x;
    const int wave  = tid >> 5;
    const int lane  = tid & 31;
    const int hi    = (lane >> 4) & 1;   // lane half
    const int ln    = lane & 15;
    const int m0    = blockIdx.x * 128 + (wave >> 1) * 16;
    const int cbase = (wave & 1) * 128;  // column half

    __shared__ alignas(16) uint16_t sB[2][DM][KC];  // 2 x 16KB, [e][k] chunk
    __shared__ float sW1[DM];
    __shared__ float sb1[DM];
    __shared__ float sb2[DM];
    if (tid < DM) {
        sW1[tid] = W1[f * DM + tid];
        sb1[tid] = b1[f * DM + tid];
        sb2[tid] = b2[f * DM + tid];
    }

    const uint16_t* w2f = W2T + (size_t)f * DM * DM;
    // low 32 bits of a generic LDS pointer == LDS byte offset
    const uint32_t lds0 = (uint32_t)(uintptr_t)(&sB[0][0][0]);

    // Stage chunk [k0, k0+32) of all 256 N-major rows into sB[buf]:
    // 1024 b128 async transfers, 2 per thread.
    auto issue_chunk = [&](int k0, int buf) {
        uint32_t lbase = lds0 + (uint32_t)buf * (DM * KC * 2);
        #pragma unroll
        for (int i = 0; i < 2; ++i) {
            int idx  = tid + i * 512;
            int e    = idx >> 2;
            int half = idx & 3;
            uint64_t g = (uint64_t)(uintptr_t)(w2f + (size_t)e * DM + k0 + half * 8);
            uint32_t l = lbase + (uint32_t)(e * (KC * 2) + half * 16);
            asm volatile("global_load_async_to_lds_b128 %0, %1, off"
                         :: "v"(l), "v"(g) : "memory");
        }
    };

    issue_chunk(0, 0);

    const float xs = x[(size_t)(m0 + ln) * NF + f];

    v8f acc[8];
    const v8f vzero = {0.f, 0.f, 0.f, 0.f, 0.f, 0.f, 0.f, 0.f};
    #pragma unroll
    for (int n = 0; n < 8; ++n) acc[n] = vzero;

    for (int kc = 0; kc < DM / KC; ++kc) {
        const int k0 = kc * KC;
        // my async copies for chunk kc done; barrier => everyone's done
        asm volatile("s_wait_asynccnt 0x0" ::: "memory");
        __syncthreads();
        if (kc + 1 < DM / KC) issue_chunk(k0 + KC, (kc + 1) & 1);

        // ---- A fragment (16-bit A 16x32 layout):
        // VGPR j holds K pair k0 + (j>=4?16:0) + 2*(j&3) + hi*8 in [15:0],[31:16]
        union { uint32_t u[8]; v16bf v; } A;
        #pragma unroll
        for (int j = 0; j < 8; ++j) {
            int kb = k0 + ((j >= 4) ? 16 : 0) + 2 * (j & 3) + hi * 8;
            float h0 = fmaxf(xs * sW1[kb]     + sb1[kb],     0.f);
            float h1 = fmaxf(xs * sW1[kb + 1] + sb1[kb + 1], 0.f);
            A.u[j] = f32_to_bf16(h0) | (f32_to_bf16(h1) << 16);
        }

        // ---- 8 B fragments from LDS + WMMA. B 32x16 layout: lane = column
        // n = lane%16; lanes 0-15: K=0..7,16..23, lanes 16-31: K=8..15,24..31
        // -> two ds b128 loads of 8 bf16 each.
        const uint16_t* bufb = &sB[kc & 1][0][0];
        #pragma unroll
        for (int n = 0; n < 8; ++n) {
            int e = cbase + n * 16 + ln;
            union { uint4 q[2]; v16bf v; } Bf;
            const uint16_t* p = bufb + e * KC + hi * 8;
            Bf.q[0] = *(const uint4*)(p);
            Bf.q[1] = *(const uint4*)(p + 16);
            acc[n] = __builtin_amdgcn_wmma_f32_16x16x32_bf16(
                false, A.v, false, Bf.v, (short)0, acc[n], false, false);
        }
        // NOTE: no tail barrier needed: buffer (kc&1) is only overwritten by
        // the issue at iteration kc+1, which follows that iteration's barrier.
    }

    // ---- store: C/D 16x16 layout: VGPR i -> M = i + hi*8, N = lane%16
    #pragma unroll
    for (int n = 0; n < 8; ++n) {
        #pragma unroll
        for (int i = 0; i < 8; ++i) {
            int m = m0 + i + hi * 8;
            int e = cbase + n * 16 + ln;
            out[((size_t)m * (NF + 1) + 1 + f) * DM + e] = acc[n][i] + sb2[e];
        }
    }
}

extern "C" void kernel_launch(void* const* d_in, const int* in_sizes, int n_in,
                              void* d_out, int out_size, void* d_ws, size_t ws_size,
                              hipStream_t stream) {
    const float* x   = (const float*)d_in[0];   // [B, 64]
    const float* W1  = (const float*)d_in[1];   // [64, 256]
    const float* b1  = (const float*)d_in[2];   // [64, 256]
    const float* W2  = (const float*)d_in[3];   // [64, 256, 256]
    const float* b2  = (const float*)d_in[4];   // [64, 256]
    const float* cls = (const float*)d_in[5];   // [256]
    float* out = (float*)d_out;                 // [B, 65, 256]

    const int nB = in_sizes[0] / NF;            // 4096
    uint16_t* W2T = (uint16_t*)d_ws;            // 64*256*256 bf16 = 8 MB

    // 1) convert+transpose W2 -> bf16 N-major
    w2_transpose_bf16<<<dim3(DM / 32, DM / 32, NF), dim3(32, 8), 0, stream>>>(W2, W2T);
    // 2) cls rows
    cls_fill<<<(nB * DM + 255) / 256, 256, 0, stream>>>(cls, out, nB);
    // 3) batched per-feature GEMM with on-the-fly A, async-LDS B staging
    feat_gemm<<<dim3(nB / 128, NF), 512, 0, stream>>>(x, W1, b1, W2T, b2, out);
}
